// LSTM_meta_50921132261825
// MI455X (gfx1250) — matrix operand order repaired and checked by reading, not compiled, not run on hardware
//
#include <hip/hip_runtime.h>

typedef __bf16 bf16_t;
typedef __attribute__((ext_vector_type(16))) __bf16 v16bf;
typedef __attribute__((ext_vector_type(8)))  __bf16 v8bf;
typedef __attribute__((ext_vector_type(8)))  float  v8f;

constexpr int Bsz = 64, Tsz = 512, Isz = 128, Hsz = 256, Gsz = 1024, Csz = 10;
constexpr int BT = Bsz * Tsz;

// ---------------------------------------------------------------- helpers ---

__device__ __forceinline__ v8f zero_v8f() {
    v8f z;
#pragma unroll
    for (int i = 0; i < 8; ++i) z[i] = 0.0f;
    return z;
}

// Async global->LDS copy (ASYNCcnt path). Generic LDS pointers carry the
// wave-relative LDS byte offset in their low 32 bits (ISA 10.2 aperture rule).
__device__ __forceinline__ void async_ld_b128(void* lds_dst, const void* gsrc) {
    uint32_t laddr = (uint32_t)(uintptr_t)lds_dst;
    asm volatile("global_load_async_to_lds_b128 %0, %1, off"
                 :: "v"(laddr), "v"(gsrc)
                 : "memory");
}
__device__ __forceinline__ void wait_async0() {
    asm volatile("s_wait_asynccnt 0x0" ::: "memory");
}

// A-fragment (16x32 bf16, row-major source with `pitch` elements per row).
// lanes 0-15 (M=lane) hold K={0..7,16..23}; lanes 16-31 hold K={8..15,24..31}.
__device__ __forceinline__ v16bf load_afrag(const bf16_t* base, int pitch) {
    int lane = threadIdx.x & 31;
    const bf16_t* p = base + (lane & 15) * pitch + ((lane >> 4) << 3);
    v8bf lo = *(const v8bf*)(p);
    v8bf hi = *(const v8bf*)(p + 16);
    return __builtin_shufflevector(lo, hi, 0,1,2,3,4,5,6,7,8,9,10,11,12,13,14,15);
}

// B-fragment (32x16). B[k][n] = W[n][k], W row-major [*, pitch].
// lane holds N = lane%16, K = {0..7,16..23} (lanes 0-15) / {8..15,24..31} (lanes 16-31).
__device__ __forceinline__ v16bf load_bfrag(const bf16_t* base, int pitch) {
    int lane = threadIdx.x & 31;
    const bf16_t* p = base + (lane & 15) * pitch + ((lane >> 4) << 3);
    v8bf lo = *(const v8bf*)(p);
    v8bf hi = *(const v8bf*)(p + 16);
    return __builtin_shufflevector(lo, hi, 0,1,2,3,4,5,6,7,8,9,10,11,12,13,14,15);
}

__device__ __forceinline__ float sigm(float x)   { return 1.0f / (1.0f + __expf(-x)); }
__device__ __forceinline__ float tanh_f(float x) { return 2.0f * sigm(2.0f * x) - 1.0f; }

// ------------------------------------------------------------- small kernels

__global__ void cvt_bf16_kernel(const float* __restrict__ s, bf16_t* __restrict__ d, int n) {
    int i = blockIdx.x * blockDim.x + threadIdx.x;
    if (i < n) d[i] = (bf16_t)s[i];
}

// x [B][T][I] f32 -> xbf [T][B][I] bf16 (transpose + convert, first layer only)
__global__ void cvt_x_kernel(const float* __restrict__ x, bf16_t* __restrict__ xbf) {
    int idx = blockIdx.x * blockDim.x + threadIdx.x;
    if (idx >= Bsz * Tsz * Isz) return;
    int i  = idx % Isz;
    int bt = idx / Isz;
    int b  = bt / Tsz, t = bt % Tsz;
    xbf[((size_t)t * Bsz + b) * Isz + i] = (bf16_t)x[idx];
}

// act = elu(out1 + out2), written as bf16 (act only feeds the next GEMM)
__global__ void elu_add_kernel(const float* __restrict__ a, const float* __restrict__ b,
                               bf16_t* __restrict__ o, int n) {
    int i = blockIdx.x * blockDim.x + threadIdx.x;
    if (i < n) {
        float v = a[i] + b[i];
        o[i] = (bf16_t)((v > 0.0f) ? v : (__expf(v) - 1.0f));
    }
}

__global__ void fc_kernel(const float* __restrict__ seq /* [T][B][H] */,
                          const float* __restrict__ w, const float* __restrict__ bias,
                          float* __restrict__ outp) {
    int idx = blockIdx.x * blockDim.x + threadIdx.x;
    if (idx >= Bsz * Csz) return;
    int b = idx / Csz, c = idx % Csz;
    const float* hrow = seq + ((size_t)(Tsz - 1) * Bsz + b) * Hsz;
    float s = bias[c];
    for (int k = 0; k < Hsz; ++k) s += hrow[k] * w[c * Hsz + k];
    outp[idx] = s;
}

// ------------------------------------------- xg = in * Wih^T + bih + bhh ----
// in: [T*B][K] bf16 (m' = t*B + b); wbf: [G][K] bf16.
// xg out: fragment-tiled f32 — tile (mt,nt) stored as 256 floats at
// ((mt*64)+nt)*256, element order = lane*8 + r (exact WMMA D-register order).
__global__ __launch_bounds__(256) void gemm_xg_kernel(
    const bf16_t* __restrict__ in, const bf16_t* __restrict__ wbf,
    const float* __restrict__ bih, const float* __restrict__ bhh,
    float* __restrict__ xg, int K)
{
    constexpr int TM = 64, TN = 256, TK = 32, LP = TK + 8;  // pitch 40 (80B, 16B-aligned)
    __shared__ bf16_t Als[TM][LP];
    __shared__ bf16_t Bls[TN][LP];

    int tid  = threadIdx.x;
    int wave = tid >> 5;
    int lane = tid & 31;
    int wm   = wave >> 2;      // 0..1
    int wn   = wave & 3;       // 0..3
    int m0   = blockIdx.x * TM;
    int n0   = blockIdx.y * TN;

    v8f acc[2][4];
#pragma unroll
    for (int i = 0; i < 2; ++i)
#pragma unroll
        for (int j = 0; j < 4; ++j) acc[i][j] = zero_v8f();

    for (int kk = 0; kk < K; kk += TK) {
        // Stage A via async copy: 64x32 bf16, one b128 per lane.
        {
            int row = wave * 8 + (lane >> 2);
            int col = (lane & 3) * 8;
            async_ld_b128(&Als[row][col], in + (size_t)(m0 + row) * K + kk + col);
        }
        // Stage B via async copy: 256x32 bf16, 4 b128 per lane (one W row).
        {
            int row = wave * 32 + lane;
            const bf16_t* src = wbf + (size_t)(n0 + row) * K + kk;
#pragma unroll
            for (int i = 0; i < 4; ++i)
                async_ld_b128(&Bls[row][i * 8], src + i * 8);
        }
        wait_async0();
        __syncthreads();

        v16bf af[2], bfr[4];
#pragma unroll
        for (int i = 0; i < 2; ++i) af[i]  = load_afrag(&Als[wm * 32 + i * 16][0], LP);
#pragma unroll
        for (int j = 0; j < 4; ++j) bfr[j] = load_bfrag(&Bls[wn * 64 + j * 16][0], LP);
#pragma unroll
        for (int i = 0; i < 2; ++i)
#pragma unroll
            for (int j = 0; j < 4; ++j)
                acc[i][j] = __builtin_amdgcn_wmma_f32_16x16x32_bf16(
                    false, af[i], false, bfr[j], (short)0, acc[i][j], false, false);
        __syncthreads();
    }

    // Epilogue: add biases, store tiles in register order (32B/lane contiguous).
    int col = lane & 15;
#pragma unroll
    for (int i = 0; i < 2; ++i)
#pragma unroll
        for (int j = 0; j < 4; ++j) {
            int n    = n0 + wn * 64 + j * 16 + col;
            float bias = bih[n] + bhh[n];
            v8f v = acc[i][j];
#pragma unroll
            for (int r = 0; r < 8; ++r) v[r] += bias;
            int mt = (m0 + wm * 32 + i * 16) >> 4;
            int nt = (n0 + wn * 64 + j * 16) >> 4;
            *(v8f*)(xg + ((size_t)mt * (Gsz / 16) + nt) * 256 + lane * 8) = v;
        }
}

// --------------------------------------------------------- LSTM recurrence -
// Grid = B/16 workgroups (batch tiles), zero cross-WG deps. Wave w owns hidden
// cols [w*32, w*32+32) across ALL four gate blocks -> i/f/g/o and the cell
// state c never leave registers; h goes to LDS (bf16) for the next step's
// A-fragments and to global in both f32 and bf16 [T][B][H] layouts.
__global__ __launch_bounds__(256) void lstm_recur_kernel(
    const float* __restrict__ xg,     // fragment-tiled (see gemm epilogue)
    const bf16_t* __restrict__ whh,   // [G][H] bf16 (L2-resident)
    float* __restrict__ out_f,        // [T][B][H] f32
    bf16_t* __restrict__ out_bf)      // [T][B][H] bf16
{
    constexpr int HP = Hsz + 8;       // pitch 264 (528B, 16B-aligned)
    __shared__ bf16_t h_lds[16][HP];

    int tid   = threadIdx.x;
    int wave  = tid >> 5;
    int lane  = tid & 31;
    int bt    = blockIdx.x;           // batch tile 0..3
    int b0    = bt * 16;
    int col   = lane & 15;
    int rbase = (lane >> 4) << 3;

    for (int i = tid; i < 16 * HP; i += 256) (&h_lds[0][0])[i] = (bf16_t)0.0f;
    v8f c[2];
    c[0] = zero_v8f();
    c[1] = zero_v8f();
    __syncthreads();

    for (int t = 0; t < Tsz; ++t) {
        // Prefetch next step's xg slab for this wave's gate columns.
        if (t + 1 < Tsz) {
            const float* nxt = xg + ((size_t)((t + 1) * 4 + bt) * (Gsz / 16) + wave * 2) * 256;
            __builtin_prefetch(nxt, 0, 3);
        }

        // A-fragments of h (16x256), one per K-step of 32.
        v16bf af[8];
#pragma unroll
        for (int ks = 0; ks < 8; ++ks) af[ks] = load_afrag(&h_lds[0][ks * 32], HP);
        __syncthreads();  // all reads of h_lds done before this step's writes

        const float* xg_t = xg + ((size_t)(t * 4 + bt) * (Gsz / 16)) * 256;
        v8f acc[4][2];
#pragma unroll
        for (int gate = 0; gate < 4; ++gate)
#pragma unroll
            for (int j = 0; j < 2; ++j) {
                int n  = gate * 256 + wave * 32 + j * 16;
                int nt = n >> 4;
                // contiguous 32B accumulator init (register-order tile)
                v8f a = *(const v8f*)(xg_t + (size_t)nt * 256 + lane * 8);
#pragma unroll
                for (int ks = 0; ks < 8; ++ks) {
                    v16bf bfr = load_bfrag(whh + (size_t)n * Hsz + ks * 32, Hsz);
                    a = __builtin_amdgcn_wmma_f32_16x16x32_bf16(
                        false, af[ks], false, bfr, (short)0, a, false, false);
                }
                acc[gate][j] = a;
            }

        // In-register gate math + state update.
#pragma unroll
        for (int j = 0; j < 2; ++j)
#pragma unroll
            for (int r = 0; r < 8; ++r) {
                float iv = sigm(acc[0][j][r]);
                float fv = sigm(acc[1][j][r]);
                float gv = tanh_f(acc[2][j][r]);
                float ov = sigm(acc[3][j][r]);
                float cn = fv * c[j][r] + iv * gv;
                c[j][r]  = cn;
                float hv = ov * tanh_f(cn);
                int m  = rbase + r;
                int hd = wave * 32 + j * 16 + col;
                bf16_t hb = (bf16_t)hv;
                h_lds[m][hd] = hb;
                size_t o = ((size_t)t * Bsz + (b0 + m)) * Hsz + hd;
                out_f[o]  = hv;
                out_bf[o] = hb;
            }
        __syncthreads();  // h_lds fully updated before next step's reads
    }
}

// ------------------------------------------------------------------ driver -

extern "C" void kernel_launch(void* const* d_in, const int* in_sizes, int n_in,
                              void* d_out, int out_size, void* d_ws, size_t ws_size,
                              hipStream_t stream) {
    const float* x      = (const float*)d_in[0];
    const float* w1_ih0 = (const float*)d_in[1];
    const float* w1_hh0 = (const float*)d_in[2];
    const float* b1_ih0 = (const float*)d_in[3];
    const float* b1_hh0 = (const float*)d_in[4];
    const float* w1_ihr = (const float*)d_in[5];
    const float* w1_hhr = (const float*)d_in[6];
    const float* b1_ihr = (const float*)d_in[7];
    const float* b1_hhr = (const float*)d_in[8];
    const float* w2_ih  = (const float*)d_in[9];
    const float* w2_hh  = (const float*)d_in[10];
    const float* b2_ih  = (const float*)d_in[11];
    const float* b2_hh  = (const float*)d_in[12];
    const float* w3_ih  = (const float*)d_in[13];
    const float* w3_hh  = (const float*)d_in[14];
    const float* b3_ih  = (const float*)d_in[15];
    const float* b3_hh  = (const float*)d_in[16];
    const float* fc_w   = (const float*)d_in[17];
    const float* fc_b   = (const float*)d_in[18];

    const size_t SEQ = (size_t)Bsz * Tsz * Hsz;
    char* ws = (char*)d_ws;
    float*  xg      = (float*)ws;                       // T*B*G f32 (tiled) = 128 MiB
    float*  seqA_f  = (float*)(ws + (size_t)Tsz * Bsz * Gsz * 4);
    float*  seqB_f  = seqA_f + SEQ;
    float*  seqC_f  = seqB_f + SEQ;
    bf16_t* seqA_bf = (bf16_t*)(seqC_f + SEQ);
    bf16_t* seqB_bf = seqA_bf + SEQ;
    bf16_t* seqC_bf = seqB_bf + SEQ;
    bf16_t* xbf     = seqC_bf + SEQ;                    // T*B*I bf16
    bf16_t* wih_bf  = xbf + (size_t)Tsz * Bsz * Isz;
    bf16_t* whh_bf  = wih_bf + (size_t)Gsz * Hsz;

    auto run_layer = [&](const bf16_t* in_bf, float* out_f, bf16_t* out_bf,
                         const float* wih, const float* whh,
                         const float* bih, const float* bhh, int K) {
        int n1 = Gsz * K;
        cvt_bf16_kernel<<<(n1 + 255) / 256, 256, 0, stream>>>(wih, wih_bf, n1);
        int n2 = Gsz * Hsz;
        cvt_bf16_kernel<<<(n2 + 255) / 256, 256, 0, stream>>>(whh, whh_bf, n2);
        dim3 grid(BT / 64, Gsz / 256);
        gemm_xg_kernel<<<grid, 256, 0, stream>>>(in_bf, wih_bf, bih, bhh, xg, K);
        lstm_recur_kernel<<<Bsz / 16, 256, 0, stream>>>(xg, whh_bf, out_f, out_bf);
    };

    const size_t GH = (size_t)Gsz * Hsz;

    // first-layer input: transpose+convert to [T][B][I] bf16
    {
        int n = Bsz * Tsz * Isz;
        cvt_x_kernel<<<(n + 255) / 256, 256, 0, stream>>>(x, xbf);
    }
    // stack 1
    run_layer(xbf,     seqA_f, seqA_bf, w1_ih0, w1_hh0, b1_ih0, b1_hh0, Isz);
    run_layer(seqA_bf, seqB_f, seqB_bf, w1_ihr, w1_hhr, b1_ihr, b1_hhr, Hsz); // out1
    // stack 2
    run_layer(seqB_bf, seqA_f, seqA_bf, w2_ih,      w2_hh,      b2_ih,       b2_hh,       Hsz);
    run_layer(seqA_bf, seqC_f, seqC_bf, w2_ih + GH, w2_hh + GH, b2_ih + Gsz, b2_hh + Gsz, Hsz); // out2
    // act = elu(out1 + out2) -> bf16 (feeds stack-3 GEMM only)
    {
        int n = (int)SEQ;
        elu_add_kernel<<<(n + 255) / 256, 256, 0, stream>>>(seqB_f, seqC_f, seqC_bf, n);
    }
    // stack 3
    run_layer(seqC_bf, seqA_f, seqA_bf, w3_ih,      w3_hh,      b3_ih,       b3_hh,       Hsz);
    run_layer(seqA_bf, seqB_f, seqB_bf, w3_ih + GH, w3_hh + GH, b3_ih + Gsz, b3_hh + Gsz, Hsz); // out3
    // FC head
    fc_kernel<<<(Bsz * Csz + 255) / 256, 256, 0, stream>>>(seqB_f, fc_w, fc_b, (float*)d_out);
}
